// MultiHeadAttention_84696755077913
// MI455X (gfx1250) — compile-verified, matrix operand order checked by reading, not run on hardware
//
#include <hip/hip_runtime.h>
#include <hip/hip_bf16.h>
#include <stdint.h>

#define E_DIM 1024
#define N_HEADS 16
#define HEAD_DIM 64
#define SEQ 2048
#define BATCH 4
#define ROWS (BATCH * SEQ)  // 8192

typedef __attribute__((ext_vector_type(16))) __bf16 v16bf;
typedef __attribute__((ext_vector_type(8)))  float  v8f;
typedef __attribute__((ext_vector_type(4)))  unsigned int u32x4;

union ABf { v16bf v; u32x4 u[2]; };

__device__ __forceinline__ unsigned short f2bf(float f) {
  union { float f; unsigned int u; } x; x.f = f;
  unsigned int r = x.u + 0x7FFFu + ((x.u >> 16) & 1u);  // round-to-nearest-even
  return (unsigned short)(r >> 16);
}

#define WMMA_BF16(A, B, C) \
  __builtin_amdgcn_wmma_f32_16x16x32_bf16(false, (A), false, (B), (short)0, (C), false, false)

// ---------------------------------------------------------------- convert
__global__ void cvt_f32_bf16(const float* __restrict__ in,
                             unsigned short* __restrict__ out, int n) {
  int i = blockIdx.x * blockDim.x + threadIdx.x;
  if (i < n) out[i] = f2bf(in[i]);
}

// ---------------------------------------------------------------- GEMM  Y = X @ W^T + bias
// X: [ROWS,1024] bf16.  W: [1024,1024] bf16 row-major (W row n == B-operand column n).
// Block tile 128x128 (8 waves = 4M x 2N), wave tile 32x64.
// Register ping-pong double buffering over K.
// mode 0: bf16 [ROWS,1024]; mode 1: f32 [ROWS,1024]; mode 2: bf16 Vt [B,H,D,S].
__global__ __launch_bounds__(256)
void gemm_xwt(const unsigned short* __restrict__ Xb,
              const unsigned short* __restrict__ Wb,
              const float* __restrict__ bias,
              void* __restrict__ outp, float scale, int mode) {
  const int lane = threadIdx.x & 31;
  const int wave = threadIdx.x >> 5;
  const int wm = wave & 3;
  const int wn = wave >> 2;
  const int mbase = blockIdx.y * 128 + wm * 32;
  const int nbase = blockIdx.x * 128 + wn * 64;
  const int lrow = lane & 15;
  const int lk = (lane >> 4) << 3;   // A/B 16-bit K-chunk base (ISA 7.12.2)
  const int half = lane >> 4;

  const u32x4* Xv = (const u32x4*)Xb;
  const u32x4* Wv = (const u32x4*)Wb;

  const int aoff0 = (mbase + lrow) * E_DIM + lk;
  const int aoff1 = aoff0 + 16 * E_DIM;
  int boff[4];
#pragma unroll
  for (int t = 0; t < 4; ++t) boff[t] = (nbase + t * 16 + lrow) * E_DIM + lk;

  v8f c[2][4] = {};
  ABf aP[2], bP[4], aN[2], bN[4];

  auto LD = [&](ABf* A, ABf* B4, int k0) {
    A[0].u[0] = Xv[(aoff0 + k0) >> 3];
    A[0].u[1] = Xv[(aoff0 + k0 + 16) >> 3];
    A[1].u[0] = Xv[(aoff1 + k0) >> 3];
    A[1].u[1] = Xv[(aoff1 + k0 + 16) >> 3];
#pragma unroll
    for (int t = 0; t < 4; ++t) {
      B4[t].u[0] = Wv[(boff[t] + k0) >> 3];
      B4[t].u[1] = Wv[(boff[t] + k0 + 16) >> 3];
    }
  };

  LD(aP, bP, 0);
#pragma unroll 1
  for (int k0 = 0; k0 < E_DIM; k0 += 64) {
    LD(aN, bN, (k0 + 32) & (E_DIM - 1));
#pragma unroll
    for (int m = 0; m < 2; ++m)
#pragma unroll
      for (int t = 0; t < 4; ++t) c[m][t] = WMMA_BF16(aP[m].v, bP[t].v, c[m][t]);
    LD(aP, bP, (k0 + 64) & (E_DIM - 1));
#pragma unroll
    for (int m = 0; m < 2; ++m)
#pragma unroll
      for (int t = 0; t < 4; ++t) c[m][t] = WMMA_BF16(aN[m].v, bN[t].v, c[m][t]);
  }

#pragma unroll
  for (int m = 0; m < 2; ++m)
#pragma unroll
    for (int t = 0; t < 4; ++t) {
      const int col = nbase + t * 16 + lrow;
      const float bv = bias[col];
#pragma unroll
      for (int r = 0; r < 8; ++r) {
        const int row = mbase + m * 16 + r + 8 * half;  // C layout: VGPR r -> M = r + 8*half
        const float v = (c[m][t][r] + bv) * scale;
        if (mode == 0) {
          ((unsigned short*)outp)[row * E_DIM + col] = f2bf(v);
        } else if (mode == 1) {
          ((float*)outp)[row * E_DIM + col] = v;
        } else {  // V transpose: [B,H,D,S]
          const int b = row >> 11, s = row & 2047;
          const int h = col >> 6, d = col & 63;
          ((unsigned short*)outp)[((b * N_HEADS + h) * HEAD_DIM + d) * SEQ + s] = f2bf(v);
        }
      }
    }
}

// ---------------------------------------------------------------- flash attention
// One wave per 32-query tile per (b,h); online softmax; V tiles prefetched
// before the softmax phase so global loads overlap the LDS work.
// Q,K: [ROWS,1024] bf16 (Q pre-scaled by 1/8).  Vt: [B,H,D,S] bf16.  Ctx: bf16 [ROWS,1024].
__global__ __launch_bounds__(32)
void attn_fwd(const unsigned short* __restrict__ Qb,
              const unsigned short* __restrict__ Kb,
              const unsigned short* __restrict__ Vtb,
              unsigned short* __restrict__ Ctx) {
  __shared__ __align__(16) float sc[32][32];
  __shared__ __align__(16) unsigned short pb[32][32];
  __shared__ float salpha[32];
  __shared__ float srecip[32];

  const int lane = threadIdx.x & 31;
  const int lrow = lane & 15;
  const int lk = (lane >> 4) << 3;
  const int half = lane >> 4;

  const int bh = blockIdx.x >> 6;   // / (SEQ/32)
  const int qt = blockIdx.x & 63;
  const int b = bh >> 4, h = bh & 15;
  const int qrowbase = b * SEQ + qt * 32;

  const u32x4* Qv = (const u32x4*)Qb;
  const u32x4* Kv = (const u32x4*)Kb;
  const u32x4* Vv = (const u32x4*)Vtb;

  // Q: 2 query sub-tiles x 2 K-dim chunks, resident in registers
  ABf aq[2][2];
#pragma unroll
  for (int m = 0; m < 2; ++m) {
    const int qoff = (qrowbase + m * 16 + lrow) * E_DIM + h * HEAD_DIM;
    aq[m][0].u[0] = Qv[(qoff + lk) >> 3];
    aq[m][0].u[1] = Qv[(qoff + lk + 16) >> 3];
    aq[m][1].u[0] = Qv[(qoff + 32 + lk) >> 3];
    aq[m][1].u[1] = Qv[(qoff + 32 + lk + 16) >> 3];
  }

  v8f o[2][4] = {};
  float m_i = -3.0e38f, l_i = 0.0f;

#pragma unroll 1
  for (int kt = 0; kt < SEQ / 32; ++kt) {
    // ---- scores: S(32x32) = Q(32x64) @ K_tile^T(64x32)
    ABf bk[2][2];
    {
      const int koff0 = (b * SEQ + kt * 32 + lrow) * E_DIM + h * HEAD_DIM;
      const int koff1 = koff0 + 16 * E_DIM;
      bk[0][0].u[0] = Kv[(koff0 + lk) >> 3];      bk[0][0].u[1] = Kv[(koff0 + lk + 16) >> 3];
      bk[0][1].u[0] = Kv[(koff0 + 32 + lk) >> 3]; bk[0][1].u[1] = Kv[(koff0 + 32 + lk + 16) >> 3];
      bk[1][0].u[0] = Kv[(koff1 + lk) >> 3];      bk[1][0].u[1] = Kv[(koff1 + lk + 16) >> 3];
      bk[1][1].u[0] = Kv[(koff1 + 32 + lk) >> 3]; bk[1][1].u[1] = Kv[(koff1 + 32 + lk + 16) >> 3];
    }
    v8f s[2][2] = {};
#pragma unroll
    for (int m = 0; m < 2; ++m)
#pragma unroll
      for (int n = 0; n < 2; ++n) {
        s[m][n] = WMMA_BF16(aq[m][0].v, bk[n][0].v, s[m][n]);
        s[m][n] = WMMA_BF16(aq[m][1].v, bk[n][1].v, s[m][n]);
      }

    // ---- prefetch V tiles now; they stay in flight through the softmax phase
    ABf bvm[4];
#pragma unroll
    for (int t = 0; t < 4; ++t) {
      const int voff = ((b * N_HEADS + h) * HEAD_DIM + t * 16 + lrow) * SEQ + kt * 32;
      bvm[t].u[0] = Vv[(voff + lk) >> 3];
      bvm[t].u[1] = Vv[(voff + lk + 16) >> 3];
    }

    // ---- score tile C-layout -> row-major in LDS
#pragma unroll
    for (int m = 0; m < 2; ++m)
#pragma unroll
      for (int r = 0; r < 8; ++r) {
        sc[m * 16 + r + 8 * half][lrow]      = s[m][0][r];
        sc[m * 16 + r + 8 * half][16 + lrow] = s[m][1][r];
      }
    __syncthreads();  // single-wave WG: S_NOP

    // ---- online softmax: every lane owns one query row
    {
      float rmax = -3.0e38f;
#pragma unroll
      for (int j = 0; j < 8; ++j) {
        const float4 v4 = *(const float4*)&sc[lane][j * 4];
        rmax = fmaxf(rmax, fmaxf(fmaxf(v4.x, v4.y), fmaxf(v4.z, v4.w)));
      }
      const float mnew = fmaxf(m_i, rmax);
      const float alpha = __expf(m_i - mnew);
      float ssum = 0.f;
#pragma unroll
      for (int j = 0; j < 16; ++j) {
        const float p0 = __expf(sc[lane][2 * j] - mnew);
        const float p1 = __expf(sc[lane][2 * j + 1] - mnew);
        ((unsigned int*)pb[lane])[j] = (unsigned int)f2bf(p0) | ((unsigned int)f2bf(p1) << 16);
        ssum += p0 + p1;
      }
      l_i = l_i * alpha + ssum;
      m_i = mnew;
      salpha[lane] = alpha;
    }
    __syncthreads();

    // ---- P back out in A-operand layout (ds_load_b128)
    ABf ap[2];
    ap[0].u[0] = *((const u32x4*)&pb[lrow][lk]);
    ap[0].u[1] = *((const u32x4*)&pb[lrow][lk + 16]);
    ap[1].u[0] = *((const u32x4*)&pb[16 + lrow][lk]);
    ap[1].u[1] = *((const u32x4*)&pb[16 + lrow][lk + 16]);

#pragma unroll
    for (int m = 0; m < 2; ++m) {
      float ar[8];
#pragma unroll
      for (int r = 0; r < 8; ++r) ar[r] = salpha[m * 16 + r + 8 * half];
#pragma unroll
      for (int t = 0; t < 4; ++t)
#pragma unroll
        for (int r = 0; r < 8; ++r) o[m][t][r] = o[m][t][r] * ar[r];
    }

    // ---- O += P(32x32) @ V_tile(32x64)
#pragma unroll
    for (int m = 0; m < 2; ++m)
#pragma unroll
      for (int t = 0; t < 4; ++t) o[m][t] = WMMA_BF16(ap[m].v, bvm[t].v, o[m][t]);
    __syncthreads();
  }

  srecip[lane] = (l_i > 0.f) ? (1.0f / l_i) : 0.f;
  __syncthreads();
#pragma unroll
  for (int m = 0; m < 2; ++m) {
    float rr[8];
#pragma unroll
    for (int r = 0; r < 8; ++r) rr[r] = srecip[m * 16 + r + 8 * half];
#pragma unroll
    for (int t = 0; t < 4; ++t) {
      const int col = h * HEAD_DIM + t * 16 + lrow;
#pragma unroll
      for (int r = 0; r < 8; ++r) {
        const int row = qrowbase + m * 16 + r + 8 * half;
        Ctx[row * E_DIM + col] = f2bf(o[m][t][r] * rr[r]);
      }
    }
  }
}

// ---------------------------------------------------------------- launch
extern "C" void kernel_launch(void* const* d_in, const int* in_sizes, int n_in,
                              void* d_out, int out_size, void* d_ws, size_t ws_size,
                              hipStream_t stream) {
  const float* x  = (const float*)d_in[0];
  const float* wq = (const float*)d_in[1];
  const float* bq = (const float*)d_in[2];
  const float* wk = (const float*)d_in[3];
  const float* bk = (const float*)d_in[4];
  const float* wv = (const float*)d_in[5];
  const float* bv = (const float*)d_in[6];
  const float* wo = (const float*)d_in[7];
  const float* bo = (const float*)d_in[8];

  char* ws = (char*)d_ws;
  const size_t MB = 1024 * 1024;
  unsigned short* xb  = (unsigned short*)(ws + 0 * MB);   // 16 MB
  unsigned short* wqb = (unsigned short*)(ws + 16 * MB);  //  2 MB
  unsigned short* wkb = (unsigned short*)(ws + 18 * MB);
  unsigned short* wvb = (unsigned short*)(ws + 20 * MB);
  unsigned short* wob = (unsigned short*)(ws + 22 * MB);
  unsigned short* qb  = (unsigned short*)(ws + 24 * MB);  // 16 MB
  unsigned short* kb2 = (unsigned short*)(ws + 40 * MB);  // 16 MB
  unsigned short* vtb = (unsigned short*)(ws + 56 * MB);  // 16 MB [B,H,D,S]
  unsigned short* ctx = (unsigned short*)(ws + 72 * MB);  // 16 MB

  const int NX = ROWS * E_DIM;
  const int NW = E_DIM * E_DIM;
  cvt_f32_bf16<<<(NX + 255) / 256, 256, 0, stream>>>(x,  xb,  NX);
  cvt_f32_bf16<<<(NW + 255) / 256, 256, 0, stream>>>(wq, wqb, NW);
  cvt_f32_bf16<<<(NW + 255) / 256, 256, 0, stream>>>(wk, wkb, NW);
  cvt_f32_bf16<<<(NW + 255) / 256, 256, 0, stream>>>(wv, wvb, NW);
  cvt_f32_bf16<<<(NW + 255) / 256, 256, 0, stream>>>(wo, wob, NW);

  dim3 gg(E_DIM / 128, ROWS / 128);  // (8, 64), 256 threads = 8 waves
  const float qscale = 0.125f;       // 1/sqrt(64)
  gemm_xwt<<<gg, 256, 0, stream>>>(xb, wqb, bq, qb,  qscale, 0);
  gemm_xwt<<<gg, 256, 0, stream>>>(xb, wkb, bk, kb2, 1.0f,   0);
  gemm_xwt<<<gg, 256, 0, stream>>>(xb, wvb, bv, vtb, 1.0f,   2);

  attn_fwd<<<BATCH * N_HEADS * (SEQ / 32), 32, 0, stream>>>(qb, kb2, vtb, ctx);

  gemm_xwt<<<gg, 256, 0, stream>>>(ctx, wob, bo, d_out, 1.0f, 1);
}
